// MultiQueryAttention_36558761624393
// MI455X (gfx1250) — compile-verified, matrix operand order checked by reading, not run on hardware
//
#include <hip/hip_runtime.h>
#include <hip/hip_bf16.h>

// ---------------------------------------------------------------------------
// MQA forward for MI455X (gfx1250): bf16 WMMA + TDM-staged LDS double buffering
//   HIDDEN=2048, S=2048, NUM_HEADS=16, HEAD_DIM=128, B=1
// ---------------------------------------------------------------------------

#define S_LEN   2048
#define HID     2048
#define NHEADS  16
#define HDIM    128

typedef __attribute__((ext_vector_type(16))) __bf16       bf16x16;
typedef __attribute__((ext_vector_type(8)))  float        f32x8;
typedef __attribute__((ext_vector_type(4)))  unsigned int u32x4;
typedef __attribute__((ext_vector_type(4)))  int          i32x4;
typedef __attribute__((ext_vector_type(8)))  int          i32x8;

#if defined(__AMDGCN__) && __has_builtin(__builtin_amdgcn_tensor_load_to_lds)
#define USE_TDM 1
#if __has_include(<hip/amd_detail/amd_gfx1250_TDM.h>)
#define TDM_SIX_ARGS 1
#endif
#endif

struct FragBF {
    union { bf16x16 v; uint4 q[2]; };
};

__device__ inline f32x8 zero8() {
    f32x8 z;
#pragma unroll
    for (int i = 0; i < 8; ++i) z[i] = 0.0f;
    return z;
}

// Load a 16x32 bf16 fragment in the CDNA5 WMMA A/B register layout (ISA 7.12.2).
//   lanes 0-15 : elems 0-7 => K k0..k0+7,   elems 8-15 => K k0+16..k0+23
//   lanes 16-31: elems 0-7 => K k0+8..k0+15, elems 8-15 => K k0+24..k0+31
// Works for global or LDS-backed (generic) pointers.
__device__ inline FragBF load_frag(const __bf16* base, int ld,
                                   int rc0, int k0, int lane) {
    const int n  = lane & 15;
    const int hf = lane >> 4;
    const __bf16* p = base + (size_t)(rc0 + n) * ld + k0 + hf * 8;
    FragBF f;
    f.q[0] = *(const uint4*)(p);
    f.q[1] = *(const uint4*)(p + 16);
    return f;
}

__device__ inline f32x8 wmma_bf16(const FragBF& a, const FragBF& b, f32x8 c) {
    return __builtin_amdgcn_wmma_f32_16x16x32_bf16(
        false, a.v, false, b.v, (short)0, c, false, false);
}

// ---------------------------------------------------------------------------
// Tensor Data Mover helpers
// ---------------------------------------------------------------------------
#ifdef USE_TDM
__device__ inline unsigned lds_offset(const void* p) {
    return (unsigned)(unsigned long long)
        (__attribute__((address_space(3))) const char*)p;
}

// 2-D tile DMA: global (row-major, strideElems between rows, bf16) -> LDS
// (packed [tileD1][tileD0] bf16). Wave-level op; issue from one wave only.
__device__ inline void tdm_load_2d(unsigned ldsOff, const void* gaddr,
                                   unsigned tileD0, unsigned tileD1,
                                   unsigned long long strideElems) {
    const unsigned long long ga = (unsigned long long)gaddr;
    u32x4 g0;
    g0[0] = 1u;                                            // count=1, load, user
    g0[1] = ldsOff;                                        // lds_addr
    g0[2] = (unsigned)(ga & 0xffffffffu);                  // global_addr[31:0]
    g0[3] = (unsigned)((ga >> 32) & 0x01ffffffu) | (2u << 30); // addr[56:32], type=2
    i32x8 g1;
    g1[0] = (int)(1u << 16);                               // wg_mask=0, data_size=1 (2B)
    g1[1] = (int)((tileD0 & 0xffffu) << 16);               // tensor_dim0[15:0]
    g1[2] = (int)(((tileD0 >> 16) & 0xffffu) | ((tileD1 & 0xffffu) << 16));
    g1[3] = (int)(((tileD1 >> 16) & 0xffffu) | ((tileD0 & 0xffffu) << 16)); // tile_dim0
    g1[4] = (int)(tileD1 & 0xffffu);                       // tile_dim1 (tile_dim2=0)
    g1[5] = (int)(unsigned)(strideElems & 0xffffffffu);    // dim0_stride[31:0]
    g1[6] = (int)(unsigned)((strideElems >> 32) & 0xffffu);// dim0_stride[47:32]
    g1[7] = 0;
    const i32x4 z4 = {0, 0, 0, 0};
#ifdef TDM_SIX_ARGS
    const i32x8 z8 = {0, 0, 0, 0, 0, 0, 0, 0};
    __builtin_amdgcn_tensor_load_to_lds(g0, g1, z4, z4, z8, 0);
#else
    __builtin_amdgcn_tensor_load_to_lds(g0, g1, z4, z4, 0);
#endif
}
#endif // USE_TDM

__device__ inline void wait_tensor0() {
#ifdef USE_TDM
#if __has_builtin(__builtin_amdgcn_s_wait_tensorcnt)
    __builtin_amdgcn_s_wait_tensorcnt(0);
#else
    asm volatile("s_wait_tensorcnt 0x0" ::: "memory");
#endif
#endif
}

// Cooperative fallback copy: 2-D bf16 tile -> packed LDS tile.
__device__ inline void coop_copy_tile(__bf16* dst, const __bf16* src,
                                      int rows, int rowElems, int ldSrc,
                                      int tid, int nthr) {
    const int perRow16 = rowElems >> 3;      // 16B units per row
    const int total = rows * perRow16;
    for (int i = tid; i < total; i += nthr) {
        const int r = i / perRow16;
        const int c = (i - r * perRow16) << 3;
        *(uint4*)(dst + (size_t)r * rowElems + c) =
            *(const uint4*)(src + (size_t)r * ldSrc + c);
    }
}

// ---------------------------------------------------------------------------
// Elementwise f32 -> bf16 convert
// ---------------------------------------------------------------------------
__global__ void cvt_bf16_kernel(const float* __restrict__ in,
                                __bf16* __restrict__ out, int n) {
    int i = blockIdx.x * blockDim.x + threadIdx.x;
    if (i < n) out[i] = (__bf16)in[i];
}

// in[R][C] f32  ->  out[C][R] bf16
__global__ void transpose_cvt_kernel(const float* __restrict__ in,
                                     __bf16* __restrict__ out, int R, int C) {
    int c = blockIdx.x * blockDim.x + threadIdx.x;
    int r = blockIdx.y;
    if (c < C) out[(size_t)c * R + r] = (__bf16)in[(size_t)r * C + c];
}

// ---------------------------------------------------------------------------
// bf16 WMMA GEMM with TDM-staged, double-buffered LDS tiles.
//   C[M][N] = A[M][K] * BT[N][K]^T + bias
// Block = 256 threads (8 waves). Block tile = 128 rows x 64 cols, K staged in
// chunks of 64. LDS: A 2x[128][64], B 2x[64][64] = 48 KiB.
// Grids are exact (no partial tiles): EXEC stays all-ones around WMMA.
// ---------------------------------------------------------------------------
__global__ __launch_bounds__(256)
void gemm_bf16_kernel(const __bf16* __restrict__ A, int lda,
                      const __bf16* __restrict__ BT, int ldb,
                      const float* __restrict__ bias,
                      float* __restrict__ outF, __bf16* __restrict__ outB,
                      int M, int N, int K, int ldout, int trans_store) {
    __shared__ __attribute__((aligned(128))) __bf16 sA[2][128 * 64];
    __shared__ __attribute__((aligned(128))) __bf16 sB[2][64 * 64];

    const int tid  = threadIdx.x;
    const int lane = tid & 31;
    const int wave = tid >> 5;
    const int rowBlk = blockIdx.y * 128;
    const int colBlk = blockIdx.x * 64;

    f32x8 acc[4];
#pragma unroll
    for (int t = 0; t < 4; ++t) acc[t] = zero8();

    // stage K-chunk k0 into buffer b
    auto stage = [&](int b, int k0) {
#ifdef USE_TDM
        if (wave == 0) {
            tdm_load_2d(lds_offset(sA[b]), A + (size_t)rowBlk * lda + k0,
                        64, 128, (unsigned long long)lda);
            tdm_load_2d(lds_offset(sB[b]), BT + (size_t)colBlk * ldb + k0,
                        64, 64, (unsigned long long)ldb);
        }
#else
        coop_copy_tile(sA[b], A + (size_t)rowBlk * lda + k0, 128, 64, lda, tid, 256);
        coop_copy_tile(sB[b], BT + (size_t)colBlk * ldb + k0, 64, 64, ldb, tid, 256);
#endif
    };

    stage(0, 0);
    wait_tensor0();
    __syncthreads();

    const int NSTAGE = K >> 6;     // K/64
    for (int s = 0; s < NSTAGE; ++s) {
        const int b = s & 1;
        if (s + 1 < NSTAGE) stage(b ^ 1, (s + 1) * 64);
#pragma unroll
        for (int j = 0; j < 2; ++j) {
            FragBF a = load_frag(sA[b], 64, wave * 16, j * 32, lane);
#pragma unroll
            for (int t = 0; t < 4; ++t) {
                FragBF bf = load_frag(sB[b], 64, t * 16, j * 32, lane);
                acc[t] = wmma_bf16(a, bf, acc[t]);
            }
        }
        wait_tensor0();
        __syncthreads();
    }

    const int hf = lane >> 4;
    const int n  = lane & 15;
#pragma unroll
    for (int t = 0; t < 4; ++t) {
        const int col = colBlk + t * 16 + n;
        const float bv = bias ? bias[col] : 0.0f;
#pragma unroll
        for (int r = 0; r < 8; ++r) {
            const int row = rowBlk + wave * 16 + hf * 8 + r;
            const float val = acc[t][r] + bv;
            if (trans_store)      outB[(size_t)col * ldout + row] = (__bf16)val;
            else if (outB)        outB[(size_t)row * ldout + col] = (__bf16)val;
            else                  outF[(size_t)row * ldout + col] = val;
        }
    }
}

// ---------------------------------------------------------------------------
// Flash attention (single KV head shared by all 16 Q heads).
//  grid = (NHEADS, S/16/8), block = 256 (8 waves, one 16-query tile each).
//  K/V tiles are TDM-staged into LDS, double buffered, shared by all 8 waves.
//  Qb [S][HID] bf16, Kb [S][HDIM] bf16, Vt [HDIM][S] bf16 -> Ob [S][HID] bf16
// ---------------------------------------------------------------------------
__global__ __launch_bounds__(256)
void mqa_attn_kernel(const __bf16* __restrict__ Qb,
                     const __bf16* __restrict__ Kb,
                     const __bf16* __restrict__ Vt,
                     __bf16* __restrict__ Ob) {
    __shared__ __attribute__((aligned(128))) __bf16 sK[2][32 * HDIM]; // [key][d]
    __shared__ __attribute__((aligned(128))) __bf16 sV[2][HDIM * 32]; // [d][key]
    __shared__ __attribute__((aligned(16)))  __bf16 sP[8][16 * 32];

    const int tid  = threadIdx.x;
    const int lane = tid & 31;
    const int wave = tid >> 5;
    const int h    = blockIdx.x;
    const int q0   = (blockIdx.y * 8 + wave) * 16;
    const int hf   = lane >> 4;
    const int m16  = lane & 15;
    const float scale = 0.08838834764831845f; // 1/sqrt(128)

    auto stage_kv = [&](int b, int kc) {
#ifdef USE_TDM
        if (wave == 0) {
            tdm_load_2d(lds_offset(sK[b]), Kb + (size_t)kc * HDIM,
                        HDIM, 32, (unsigned long long)HDIM);
            tdm_load_2d(lds_offset(sV[b]), Vt + kc,
                        32, HDIM, (unsigned long long)S_LEN);
        }
#else
        coop_copy_tile(sK[b], Kb + (size_t)kc * HDIM, 32, HDIM, HDIM, tid, 256);
        coop_copy_tile(sV[b], Vt + kc, HDIM, 32, S_LEN, tid, 256);
#endif
    };

    // Resident Q fragments: 16 rows x 128 (4 k-chunks of 32)
    FragBF qf[4];
#pragma unroll
    for (int j = 0; j < 4; ++j)
        qf[j] = load_frag(Qb, HID, q0, h * HDIM + j * 32, lane);

    f32x8 acc[8];
#pragma unroll
    for (int t = 0; t < 8; ++t) acc[t] = zero8();
    float mrow[8], lrow[8];
#pragma unroll
    for (int r = 0; r < 8; ++r) { mrow[r] = -1e30f; lrow[r] = 0.0f; }

    __bf16* pl = sP[wave];

    stage_kv(0, 0);
    wait_tensor0();
    __syncthreads();

    for (int kc = 0; kc < S_LEN; kc += 32) {
        const int b = (kc >> 5) & 1;
        if (kc + 32 < S_LEN) stage_kv(b ^ 1, kc + 32);

        // ---- scores: S[16 x 32] = Q[16 x 128] * K_tile^T (from LDS) ----
        f32x8 s0 = zero8(), s1 = zero8();
#pragma unroll
        for (int j = 0; j < 4; ++j) {
            FragBF kb0 = load_frag(sK[b], HDIM, 0,  j * 32, lane);
            FragBF kb1 = load_frag(sK[b], HDIM, 16, j * 32, lane);
            s0 = wmma_bf16(qf[j], kb0, s0);
            s1 = wmma_bf16(qf[j], kb1, s1);
        }
        // ---- online softmax ----
        float rmax[8];
#pragma unroll
        for (int r = 0; r < 8; ++r) {
            s0[r] *= scale; s1[r] *= scale;
            rmax[r] = fmaxf(s0[r], s1[r]);
        }
#pragma unroll
        for (int off = 1; off < 16; off <<= 1)
#pragma unroll
            for (int r = 0; r < 8; ++r)
                rmax[r] = fmaxf(rmax[r], __shfl_xor(rmax[r], off, 32));
        float corr[8], rsum[8];
#pragma unroll
        for (int r = 0; r < 8; ++r) {
            const float nm = fmaxf(mrow[r], rmax[r]);
            corr[r] = __expf(mrow[r] - nm);
            mrow[r] = nm;
            const float p0 = __expf(s0[r] - nm);
            const float p1 = __expf(s1[r] - nm);
            s0[r] = p0; s1[r] = p1;
            rsum[r] = p0 + p1;
        }
#pragma unroll
        for (int off = 1; off < 16; off <<= 1)
#pragma unroll
            for (int r = 0; r < 8; ++r)
                rsum[r] += __shfl_xor(rsum[r], off, 32);
#pragma unroll
        for (int r = 0; r < 8; ++r) lrow[r] = lrow[r] * corr[r] + rsum[r];
#pragma unroll
        for (int t = 0; t < 8; ++t)
#pragma unroll
            for (int r = 0; r < 8; ++r) acc[t][r] *= corr[r];

        // ---- transpose P (C-layout -> A-layout), wave-private LDS round trip.
        // DS ops are in-order within a wave; no barrier needed.
#pragma unroll
        for (int r = 0; r < 8; ++r) {
            const int row = hf * 8 + r;
            pl[row * 32 + m16]      = (__bf16)s0[r];
            pl[row * 32 + 16 + m16] = (__bf16)s1[r];
        }
        FragBF pf;
        {
            const __bf16* pp = pl + m16 * 32 + hf * 8;
            pf.q[0] = *(const uint4*)(pp);
            pf.q[1] = *(const uint4*)(pp + 16);
        }
        // ---- acc += P[16 x 32] * V_tile[32 x 128] (Vt tile in LDS [d][key]) ----
#pragma unroll
        for (int t = 0; t < 8; ++t) {
            FragBF vb = load_frag(sV[b], 32, t * 16, 0, lane);
            acc[t] = wmma_bf16(pf, vb, acc[t]);
        }

        wait_tensor0();
        __syncthreads();
    }

    // ---- epilogue: divide by row sums, write O in bf16 ----
#pragma unroll
    for (int t = 0; t < 8; ++t)
#pragma unroll
        for (int r = 0; r < 8; ++r) {
            const int row = q0 + hf * 8 + r;
            const int col = h * HDIM + t * 16 + m16;
            Ob[(size_t)row * HID + col] = (__bf16)(acc[t][r] / lrow[r]);
        }
}

// ---------------------------------------------------------------------------
// Host-side launcher
// ---------------------------------------------------------------------------
extern "C" void kernel_launch(void* const* d_in, const int* in_sizes, int n_in,
                              void* d_out, int out_size, void* d_ws, size_t ws_size,
                              hipStream_t stream) {
    const float* x  = (const float*)d_in[0];
    const float* Wq = (const float*)d_in[1];
    const float* bq = (const float*)d_in[2];
    const float* Wk = (const float*)d_in[3];
    const float* bk = (const float*)d_in[4];
    const float* Wv = (const float*)d_in[5];
    const float* bv = (const float*)d_in[6];
    const float* Wo = (const float*)d_in[7];
    const float* bo = (const float*)d_in[8];
    float* out = (float*)d_out;

    char* ws = (char*)d_ws;
    const size_t MB = 1ull << 20;
    __bf16* xb  = (__bf16*)(ws + 0 * MB);             // [S][HID]      8 MiB
    __bf16* WqT = (__bf16*)(ws + 8 * MB);             // [HID][HID]    8 MiB
    __bf16* WkT = (__bf16*)(ws + 16 * MB);            // [HDIM][HID]   0.5 MiB
    __bf16* WvT = (__bf16*)(ws + 16 * MB + 512*1024); // [HDIM][HID]   0.5 MiB
    __bf16* WoT = (__bf16*)(ws + 17 * MB);            // [HID][HID]    8 MiB
    __bf16* Qb  = (__bf16*)(ws + 25 * MB);            // [S][HID]      8 MiB
    __bf16* Kb  = (__bf16*)(ws + 33 * MB);            // [S][HDIM]     0.5 MiB
    __bf16* Vt  = (__bf16*)(ws + 33 * MB + 512*1024); // [HDIM][S]     0.5 MiB
    __bf16* Ob  = (__bf16*)(ws + 34 * MB);            // [S][HID]      8 MiB

    // --- convert inputs to bf16, weights transposed to [N][K] ---
    cvt_bf16_kernel<<<(S_LEN * HID + 255) / 256, 256, 0, stream>>>(x, xb, S_LEN * HID);
    transpose_cvt_kernel<<<dim3(HID / 256, HID), 256, 0, stream>>>(Wq, WqT, HID, HID);
    transpose_cvt_kernel<<<dim3(1, HID),         256, 0, stream>>>(Wk, WkT, HID, HDIM);
    transpose_cvt_kernel<<<dim3(1, HID),         256, 0, stream>>>(Wv, WvT, HID, HDIM);
    transpose_cvt_kernel<<<dim3(HID / 256, HID), 256, 0, stream>>>(Wo, WoT, HID, HID);

    // --- projections ---
    gemm_bf16_kernel<<<dim3(HID / 64, S_LEN / 128), 256, 0, stream>>>(
        xb, HID, WqT, HID, bq, nullptr, Qb, S_LEN, HID, HID, HID, 0);
    gemm_bf16_kernel<<<dim3(HDIM / 64, S_LEN / 128), 256, 0, stream>>>(
        xb, HID, WkT, HID, bk, nullptr, Kb, S_LEN, HDIM, HID, HDIM, 0);
    gemm_bf16_kernel<<<dim3(HDIM / 64, S_LEN / 128), 256, 0, stream>>>(
        xb, HID, WvT, HID, bv, nullptr, Vt, S_LEN, HDIM, HID, S_LEN, 1);

    // --- attention ---
    mqa_attn_kernel<<<dim3(NHEADS, S_LEN / 16 / 8), 256, 0, stream>>>(Qb, Kb, Vt, Ob);

    // --- output projection (f32 result) ---
    gemm_bf16_kernel<<<dim3(HID / 64, S_LEN / 128), 256, 0, stream>>>(
        Ob, HID, WoT, HID, bo, out, nullptr, S_LEN, HID, HID, HID, 0);
}
